// DistanceAwareAttention_23742579212686
// MI455X (gfx1250) — compile-verified, matrix operand order checked by reading, not run on hardware
//
#include <hip/hip_runtime.h>
#include <hip/hip_bf16.h>

#define B_  8
#define N_  1024
#define H_  512
#define NH_ 8
#define HD_ 64

typedef __bf16 bf16_t;
typedef __attribute__((ext_vector_type(16))) __bf16 v16bf;
typedef __attribute__((ext_vector_type(8)))  __bf16 v8bf;
typedef __attribute__((ext_vector_type(4)))  __bf16 v4bf;
typedef __attribute__((ext_vector_type(8)))  float  v8f;

__device__ __forceinline__ v16bf make_frag(v8bf lo, v8bf hi) {
    v16bf r;
#pragma unroll
    for (int i = 0; i < 8; ++i) { r[i] = lo[i]; r[i + 8] = hi[i]; }
    return r;
}

// Load a 16-element bf16 WMMA fragment: 8 contiguous at p, 8 contiguous at p+16.
// Caller has already folded in kbase = (lane<16 ? 0 : 8).
__device__ __forceinline__ v16bf load_frag_g(const bf16_t* p) {
    v8bf lo = *(const v8bf*)(p);
    v8bf hi = *(const v8bf*)(p + 16);
    return make_frag(lo, hi);
}

__device__ __forceinline__ v8f wmma_bf16(v16bf a, v16bf b, v8f c) {
    return __builtin_amdgcn_wmma_f32_16x16x32_bf16(false, a, false, b, (short)0, c, false, false);
}

// CDNA5 async LDS copy: 16 bytes global -> LDS, tracked by ASYNCcnt.
// VDST = per-lane LDS byte offset, VADDR = per-lane 64-bit global address.
__device__ __forceinline__ void async_cp16(unsigned lds_off, unsigned long long gaddr) {
    asm volatile("global_load_async_to_lds_b128 %0, %1, off"
                 :: "v"(lds_off), "v"(gaddr) : "memory");
}

__device__ __forceinline__ void wait_asynccnt_le4() {
    asm volatile("s_wait_asynccnt 0x4" ::: "memory");
}

// ---------------------------------------------------------------- f32 -> bf16
__global__ void cvt_f32_bf16(const float* __restrict__ in, bf16_t* __restrict__ out, int n) {
    int i = (blockIdx.x * blockDim.x + threadIdx.x) * 4;
    if (i + 3 < n) {
        float4 v = *(const float4*)(in + i);
        v4bf p;
        p[0] = (bf16_t)v.x; p[1] = (bf16_t)v.y; p[2] = (bf16_t)v.z; p[3] = (bf16_t)v.w;
        *(v4bf*)(out + i) = p;
    } else {
        for (; i < n; ++i) out[i] = (bf16_t)in[i];
    }
}

// ---------------------------------------------------------------- QKV GEMM
// xb [B*N][H] bf16 ; w [H_out][H_in] bf16 (row = B-frag column, contiguous K)
// z: 0->Q [B][NH][N][HD], 1->K [B][NH][N][HD], 2->V transposed [B][NH][HD][N]
__global__ __launch_bounds__(256) void qkv_gemm(
    const bf16_t* __restrict__ xb,
    const bf16_t* __restrict__ wq, const bf16_t* __restrict__ wk, const bf16_t* __restrict__ wv,
    const float* __restrict__ bq, const float* __restrict__ bk, const float* __restrict__ bv,
    bf16_t* __restrict__ Qb, bf16_t* __restrict__ Kb, bf16_t* __restrict__ Vt)
{
    const int lane = threadIdx.x & 31;
    const int wave = threadIdx.x >> 5;
    const int tm   = blockIdx.x * 8 + wave;  // 16-row tile id, 0..511
    const int tn4  = blockIdx.y;             // 64-col tile id, 0..7
    const int z    = blockIdx.z;             // 0=Q 1=K 2=V

    const bf16_t* w    = (z == 0) ? wq : (z == 1) ? wk : wv;
    const float*  bias = (z == 0) ? bq : (z == 1) ? bk : bv;

    const int l15   = lane & 15;
    const int kbase = (lane < 16) ? 0 : 8;
    const bf16_t* ap = xb + (size_t)(tm * 16 + l15) * H_;

    const bf16_t* bp[4];
#pragma unroll
    for (int s = 0; s < 4; ++s)
        bp[s] = w + (size_t)(tn4 * 64 + s * 16 + l15) * H_;

    v8f acc[4] = {};
    for (int kk = 0; kk < H_; kk += 32) {
        __builtin_prefetch(ap + kk + 64, 0, 0);
        v16bf a = load_frag_g(ap + kk + kbase);
#pragma unroll
        for (int s = 0; s < 4; ++s) {
            v16bf bfr = load_frag_g(bp[s] + kk + kbase);
            acc[s] = wmma_bf16(a, bfr, acc[s]);
        }
    }

    // C layout: row = tm*16 + i + 8*(lane>=16)  (8 consecutive tokens), col = tile + l15
    const int token0 = tm * 16 + (lane >> 4) * 8;
    const int bb = token0 >> 10;       // tile never crosses a batch boundary
    const int n0 = token0 & (N_ - 1);

#pragma unroll
    for (int s = 0; s < 4; ++s) {
        const int col = tn4 * 64 + s * 16 + l15;
        const int hh = col >> 6, dd = col & 63;
        const float bv_ = bias[col];
        if (z == 2) {  // V transposed: 8 consecutive n -> packed 16B store
            v8bf pk;
#pragma unroll
            for (int i = 0; i < 8; ++i) pk[i] = (bf16_t)(acc[s][i] + bv_);
            *(v8bf*)(Vt + ((size_t)(bb * NH_ + hh) * HD_ + dd) * N_ + n0) = pk;
        } else {
            bf16_t* dst = (z == 0) ? Qb : Kb;
#pragma unroll
            for (int i = 0; i < 8; ++i)
                dst[((size_t)(bb * NH_ + hh) * N_ + (n0 + i)) * HD_ + dd] =
                    (bf16_t)(acc[s][i] + bv_);
        }
    }
}

// ---------------------------------------------------------------- attention
// 4 waves per block: 4 query tiles sharing one (b,h); K/V tiles staged into
// LDS with async-to-LDS copies, double buffered; flash softmax over 32-key steps.
__global__ __launch_bounds__(128) void attn_kernel(
    const bf16_t* __restrict__ Qb, const bf16_t* __restrict__ Kb, const bf16_t* __restrict__ Vt,
    const float* __restrict__ dmat, const unsigned char* __restrict__ mask,
    const float* __restrict__ dist_emb, const float* __restrict__ attn_bias,
    bf16_t* __restrict__ aout)
{
    __shared__ __align__(16) bf16_t kbuf[2][32 * HD_];   // [key][d]   4KB each
    __shared__ __align__(16) bf16_t vbuf[2][HD_ * 32];   // [d][key]   4KB each
    __shared__ float lds_p[4][16 * 32];                  // per-wave P reshape

    const int tid  = threadIdx.x;
    const int lane = tid & 31;
    const int wv_  = tid >> 5;
    const int qt = blockIdx.x * 4 + wv_;   // 16-query tile, 0..63
    const int h  = blockIdx.y;             // 0..7
    const int b  = blockIdx.z;             // 0..7

    const int l15   = lane & 15;
    const int half  = lane >> 4;
    const int kbase = half ? 8 : 0;

    const bf16_t* Qrow = Qb + ((size_t)(b * NH_ + h) * N_ + qt * 16 + l15) * HD_;
    const v16bf qa0 = load_frag_g(Qrow + kbase);        // d in [0,32)
    const v16bf qa1 = load_frag_g(Qrow + 32 + kbase);   // d in [32,64)

    const char* Kpan = (const char*)(Kb + (size_t)(b * NH_ + h) * N_ * HD_);
    const char* Vpan = (const char*)(Vt + (size_t)(b * NH_ + h) * HD_ * N_);
    const float abias = attn_bias[h];

    // stage K/V tile `it` into buffer `buf`: 8KB total, 16B per async op,
    // 4 async ops per lane-slot across the 128 threads (4 per wave).
    auto stage = [&](int it, int buf) {
        const unsigned klds = (unsigned)(size_t)&kbuf[buf][0];
        const unsigned vlds = (unsigned)(size_t)&vbuf[buf][0];
        const char* kg = Kpan + (size_t)it * 32 * HD_ * 2;   // 4KB contiguous
#pragma unroll
        for (int r = 0; r < 2; ++r) {
            const int c = tid + r * 128;                      // 16B chunk id, 0..255
            async_cp16(klds + c * 16, (unsigned long long)(size_t)(kg + c * 16));
            const int d = c >> 2, s = c & 3;                  // V: row d, 16B sub-chunk s
            async_cp16(vlds + (unsigned)(d * 64 + s * 16),
                       (unsigned long long)(size_t)(Vpan + (size_t)d * (N_ * 2) + it * 64 + s * 16));
        }
    };

    float mrow[8], ssum[8];
    v8f o[4] = {};
#pragma unroll
    for (int i = 0; i < 8; ++i) { mrow[i] = -1e30f; ssum[i] = 0.f; }

    const int NIT = N_ / 32;
    stage(0, 0);
    stage(1, 1);

    for (int it = 0; it < NIT; ++it) {
        const int cur = it & 1;
        wait_asynccnt_le4();     // our 4 ops for tile `it` done (only next tile pending)
        __syncthreads();         // all waves' quarters landed in LDS

        // ---- scores: 16q x 32k from LDS K tile
        const bf16_t* kcur = kbuf[cur];
        v8f s0 = {}, s1 = {};
        {
            const bf16_t* kr0 = kcur + l15 * HD_;
            const bf16_t* kr1 = kcur + (16 + l15) * HD_;
            s0 = wmma_bf16(qa0, load_frag_g(kr0 + kbase), s0);
            s0 = wmma_bf16(qa1, load_frag_g(kr0 + 32 + kbase), s0);
            s1 = wmma_bf16(qa0, load_frag_g(kr1 + kbase), s1);
            s1 = wmma_bf16(qa1, load_frag_g(kr1 + 32 + kbase), s1);
        }

        // ---- scale + distance-bin bias + key mask (elementwise in C layout)
        const int c0 = it * 32 + l15, c1 = c0 + 16;
        const bool m0 = mask[b * N_ + c0] != 0;
        const bool m1 = mask[b * N_ + c1] != 0;
        float nm[8];
#pragma unroll
        for (int i = 0; i < 8; ++i) {
            const int r = qt * 16 + i + half * 8;
            const float* dr = dmat + ((size_t)b * N_ + r) * N_;
            int b0 = (int)(dr[c0] * 10.f); b0 = b0 < 0 ? 0 : (b0 > 49 ? 49 : b0);
            int b1 = (int)(dr[c1] * 10.f); b1 = b1 < 0 ? 0 : (b1 > 49 ? 49 : b1);
            float v0 = m0 ? -1e9f : (s0[i] * 0.125f + dist_emb[b0 * NH_ + h] + abias);
            float v1 = m1 ? -1e9f : (s1[i] * 0.125f + dist_emb[b1 * NH_ + h] + abias);
            s0[i] = v0; s1[i] = v1;
            nm[i] = fmaxf(v0, v1);
        }

        // ---- online softmax: row reductions across the 16-lane column group
        float corr[8];
#pragma unroll
        for (int i = 0; i < 8; ++i) {
            float v = nm[i];
            v = fmaxf(v, __shfl_xor(v, 1, 32));
            v = fmaxf(v, __shfl_xor(v, 2, 32));
            v = fmaxf(v, __shfl_xor(v, 4, 32));
            v = fmaxf(v, __shfl_xor(v, 8, 32));
            v = fmaxf(v, mrow[i]);
            corr[i] = __expf(mrow[i] - v);
            mrow[i] = v;
            float p0 = __expf(s0[i] - v);
            float p1 = __expf(s1[i] - v);
            s0[i] = p0; s1[i] = p1;
            float rs = p0 + p1;
            rs += __shfl_xor(rs, 1, 32);
            rs += __shfl_xor(rs, 2, 32);
            rs += __shfl_xor(rs, 4, 32);
            rs += __shfl_xor(rs, 8, 32);
            ssum[i] = ssum[i] * corr[i] + rs;
        }
#pragma unroll
        for (int t = 0; t < 4; ++t)
#pragma unroll
            for (int i = 0; i < 8; ++i) o[t][i] *= corr[i];

        // ---- reshape P (C layout -> A fragment) via per-wave LDS region
#pragma unroll
        for (int i = 0; i < 8; ++i) {
            const int rloc = i + half * 8;
            lds_p[wv_][rloc * 32 + l15]      = s0[i];
            lds_p[wv_][rloc * 32 + 16 + l15] = s1[i];
        }
        __syncthreads();   // fence P stores vs P loads (also uniform across waves)
        v16bf pa;
        {
            const float* pr = &lds_p[wv_][l15 * 32];
#pragma unroll
            for (int e = 0; e < 8; ++e) {
                pa[e]     = (bf16_t)pr[kbase + e];
                pa[e + 8] = (bf16_t)pr[16 + kbase + e];
            }
        }

        // ---- O += P @ V  from LDS V tile ([d][key], contiguous in key dim)
        const bf16_t* vcur = vbuf[cur];
#pragma unroll
        for (int t = 0; t < 4; ++t) {
            const bf16_t* vr = vcur + (t * 16 + l15) * 32;
            o[t] = wmma_bf16(pa, load_frag_g(vr + kbase), o[t]);
        }

        __syncthreads();   // everyone done reading buffer `cur`
        if (it + 2 < NIT) stage(it + 2, cur);   // overwrite it safely, async
    }

    // ---- normalize and store bf16 [B][N][H]
#pragma unroll
    for (int t = 0; t < 4; ++t) {
        const int d = t * 16 + l15;
#pragma unroll
        for (int i = 0; i < 8; ++i) {
            const int r = qt * 16 + i + half * 8;
            aout[(size_t)(b * N_ + r) * H_ + h * HD_ + d] = (bf16_t)(o[t][i] / ssum[i]);
        }
    }
}

// ---------------------------------------------------------------- output GEMM
__global__ __launch_bounds__(256) void out_gemm(
    const bf16_t* __restrict__ ab, const bf16_t* __restrict__ wo,
    const float* __restrict__ bo, float* __restrict__ out)
{
    const int lane = threadIdx.x & 31;
    const int wave = threadIdx.x >> 5;
    const int tm   = blockIdx.x * 8 + wave;
    const int tn4  = blockIdx.y;
    const int l15  = lane & 15;
    const int kbase = (lane < 16) ? 0 : 8;

    const bf16_t* ap = ab + (size_t)(tm * 16 + l15) * H_;
    const bf16_t* bp[4];
#pragma unroll
    for (int s = 0; s < 4; ++s)
        bp[s] = wo + (size_t)(tn4 * 64 + s * 16 + l15) * H_;

    v8f acc[4] = {};
    for (int kk = 0; kk < H_; kk += 32) {
        __builtin_prefetch(ap + kk + 64, 0, 0);
        v16bf a = load_frag_g(ap + kk + kbase);
#pragma unroll
        for (int s = 0; s < 4; ++s)
            acc[s] = wmma_bf16(a, load_frag_g(bp[s] + kk + kbase), acc[s]);
    }

    const int row0 = tm * 16 + (lane >> 4) * 8;
#pragma unroll
    for (int s = 0; s < 4; ++s) {
        const int col = tn4 * 64 + s * 16 + l15;
        const float bo_ = bo[col];
#pragma unroll
        for (int i = 0; i < 8; ++i)
            out[(size_t)(row0 + i) * H_ + col] = acc[s][i] + bo_;
    }
}

// ----------------------------------------------------------------------------
extern "C" void kernel_launch(void* const* d_in, const int* in_sizes, int n_in,
                              void* d_out, int out_size, void* d_ws, size_t ws_size,
                              hipStream_t stream)
{
    const float* x    = (const float*)d_in[0];
    const float* dmat = (const float*)d_in[1];
    const unsigned char* mask = (const unsigned char*)d_in[2];  // jax bool = 1 byte
    const float* Wq = (const float*)d_in[3];
    const float* bq = (const float*)d_in[4];
    const float* Wk = (const float*)d_in[5];
    const float* bk = (const float*)d_in[6];
    const float* Wv = (const float*)d_in[7];
    const float* bv = (const float*)d_in[8];
    const float* Wo = (const float*)d_in[9];
    const float* bo = (const float*)d_in[10];
    const float* dist_emb  = (const float*)d_in[11];
    const float* attn_bias = (const float*)d_in[12];
    float* out = (float*)d_out;

    char* ws = (char*)d_ws;
    size_t off = 0;
    const size_t NX = (size_t)B_ * N_ * H_;   // 4M elements
    const size_t NW = (size_t)H_ * H_;        // 256K elements
    const size_t NQ = (size_t)B_ * NH_ * N_ * HD_;

    bf16_t* xb  = (bf16_t*)(ws + off); off += NX * 2;
    bf16_t* wqb = (bf16_t*)(ws + off); off += NW * 2;
    bf16_t* wkb = (bf16_t*)(ws + off); off += NW * 2;
    bf16_t* wvb = (bf16_t*)(ws + off); off += NW * 2;
    bf16_t* wob = (bf16_t*)(ws + off); off += NW * 2;
    bf16_t* Qb  = (bf16_t*)(ws + off); off += NQ * 2;
    bf16_t* Kb  = (bf16_t*)(ws + off); off += NQ * 2;
    bf16_t* Vt  = (bf16_t*)(ws + off); off += NQ * 2;
    bf16_t* aob = (bf16_t*)(ws + off); off += NX * 2;

    // 1) precision conversion
    cvt_f32_bf16<<<(int)((NX / 4 + 255) / 256), 256, 0, stream>>>(x,  xb,  (int)NX);
    cvt_f32_bf16<<<(int)((NW / 4 + 255) / 256), 256, 0, stream>>>(Wq, wqb, (int)NW);
    cvt_f32_bf16<<<(int)((NW / 4 + 255) / 256), 256, 0, stream>>>(Wk, wkb, (int)NW);
    cvt_f32_bf16<<<(int)((NW / 4 + 255) / 256), 256, 0, stream>>>(Wv, wvb, (int)NW);
    cvt_f32_bf16<<<(int)((NW / 4 + 255) / 256), 256, 0, stream>>>(Wo, wob, (int)NW);

    // 2) fused QKV projections (z selects Q/K/V)
    qkv_gemm<<<dim3(64, 8, 3), 256, 0, stream>>>(xb, wqb, wkb, wvb, bq, bk, bv, Qb, Kb, Vt);

    // 3) flash attention: async-LDS staged K/V, distance-bin bias + key mask
    attn_kernel<<<dim3(N_ / 16 / 4, NH_, B_), 128, 0, stream>>>(
        Qb, Kb, Vt, dmat, mask, dist_emb, attn_bias, aob);

    // 4) output projection -> f32
    out_gemm<<<dim3(64, 8, 1), 256, 0, stream>>>(aob, wob, bo, out);
}
